// BRNN_5686536700315
// MI455X (gfx1250) — compile-verified, matrix operand order checked by reading, not run on hardware
//
#include <hip/hip_runtime.h>

#define T_DIM 2048
#define B_DIM 128
#define I_DIM 128
#define H_DIM 256
#define O_DIM 128
#define BH (B_DIM * H_DIM)
#define NWG 64

typedef __attribute__((ext_vector_type(16))) _Float16 v16h;
typedef __attribute__((ext_vector_type(8)))  _Float16 v8h;
typedef __attribute__((ext_vector_type(8)))  float    v8f;

// ---------------------------------------------------------------------------
// WMMA helper: D = A(16x32 f16) * B(32x16 f16) + C(16x16 f32)
// ---------------------------------------------------------------------------
__device__ __forceinline__ v8f wmma32(v16h a, v16h b, v8f c) {
  return __builtin_amdgcn_wmma_f32_16x16x32_f16(
      /*neg_a=*/false, a, /*neg_b=*/false, b,
      /*c_mod=*/(short)0, c, /*reuse_a=*/false, /*reuse_b=*/false);
}

// Per-lane 16 halves of an A/B fragment (ISA 7.12.2 16-bit layout):
// lane owns row/col (lane&15); k-base = (lane>>4)*8; elems 0-7 -> K kb..kb+7,
// elems 8-15 -> K kb+16..kb+23.  Two contiguous 16B loads.
__device__ __forceinline__ v16h frag_h16(const _Float16* p) {
  v8h lo = *(const v8h*)(p);
  v8h hi = *(const v8h*)(p + 16);
  v16h v;
  #pragma unroll
  for (int i = 0; i < 8; ++i) { v[i] = lo[i]; v[i + 8] = hi[i]; }
  return v;
}

// Fragment from f16 row-major matrix [rows, ldk] (global or LDS pointer).
__device__ __forceinline__ v16h frag_rm(const _Float16* M, int ldk, int row0,
                                        int k0, int lane) {
  int r  = row0 + (lane & 15);
  int kb = k0 + ((lane >> 4) << 3);
  return frag_h16(M + (size_t)r * ldk + kb);
}

// A-fragment straight from f32 input x[B][T][I] at time ts (cvt f32->f16 inline)
__device__ __forceinline__ v16h frag_x(const float* x, int ts, int row0, int k0,
                                       int lane) {
  int r  = row0 + (lane & 15);
  int kb = k0 + ((lane >> 4) << 3);
  const float* p = x + ((size_t)r * T_DIM + ts) * I_DIM + kb;
  v16h v;
  #pragma unroll
  for (int i = 0; i < 8; ++i) {
    v[i]     = (_Float16)p[i];
    v[i + 8] = (_Float16)p[16 + i];
  }
  return v;
}

// Prefetch next timestep's x fragment lines (global_prefetch_b8)
__device__ __forceinline__ void prefetch_x(const float* x, int ts, int row0,
                                           int k0, int lane) {
  int r  = row0 + (lane & 15);
  int kb = k0 + ((lane >> 4) << 3);
  const float* p = x + ((size_t)r * T_DIM + ts) * I_DIM + kb;
  __builtin_prefetch(p, 0, 1);
}

// Fast transcendentals on HW trans units (v_exp, v_rcp)
__device__ __forceinline__ float sigm(float xv) {
  return __builtin_amdgcn_rcpf(1.0f + __builtin_amdgcn_exp2f(-1.44269504f * xv));
}
__device__ __forceinline__ float tanhx(float xv) {
  float e = __builtin_amdgcn_exp2f(2.88539008f * xv);  // exp(2x)
  return 1.0f - 2.0f * __builtin_amdgcn_rcpf(e + 1.0f);
}

// ---------------------------------------------------------------------------
// Grid-wide barrier (generation counter), NWG workgroups, all co-resident.
// ---------------------------------------------------------------------------
__device__ __forceinline__ void grid_sync(int* cnt, volatile int* gen) {
  __threadfence();     // release: flush this thread's stores to device scope
  __syncthreads();
  if (threadIdx.x == 0) {
    int g = *gen;
    if (atomicAdd(cnt, 1) == NWG - 1) {
      *cnt = 0;
      __threadfence();
      atomicAdd((int*)gen, 1);
    } else {
      while (*gen == g) { __builtin_amdgcn_s_sleep(2); }
    }
  }
  __syncthreads();
  __threadfence();     // acquire: invalidate stale near-cache lines
}

// ---------------------------------------------------------------------------
// Pre-pass: convert weights f32->f16 into workspace, zero recurrent state.
// ---------------------------------------------------------------------------
__global__ void brnn_init(const float* __restrict__ w1x, const float* __restrict__ w1h,
                          const float* __restrict__ w2x, const float* __restrict__ w2h,
                          const float* __restrict__ wo,
                          _Float16* __restrict__ W1X, _Float16* __restrict__ W1H,
                          _Float16* __restrict__ W2X, _Float16* __restrict__ W2H,
                          _Float16* __restrict__ WO,
                          _Float16* __restrict__ H1H, _Float16* __restrict__ H2H,
                          float* __restrict__ C1, float* __restrict__ H2,
                          int* __restrict__ syncp) {
  int i = blockIdx.x * blockDim.x + threadIdx.x;
  int stride = gridDim.x * blockDim.x;
  for (int k = i; k < 4 * H_DIM * I_DIM; k += stride) W1X[k] = (_Float16)w1x[k];
  for (int k = i; k < 4 * H_DIM * H_DIM; k += stride) W1H[k] = (_Float16)w1h[k];
  for (int k = i; k < 3 * H_DIM * H_DIM; k += stride) {
    W2X[k] = (_Float16)w2x[k];
    W2H[k] = (_Float16)w2h[k];
  }
  for (int k = i; k < O_DIM * 2 * H_DIM; k += stride) WO[k] = (_Float16)wo[k];
  for (int k = i; k < 4 * BH; k += stride) {  // 2 parity x 2 dir
    H1H[k] = (_Float16)0.0f;
    H2H[k] = (_Float16)0.0f;
  }
  for (int k = i; k < 2 * BH; k += stride) {  // 2 dir
    C1[k] = 0.0f;
    H2[k] = 0.0f;
  }
  if (i == 0) { syncp[0] = 0; syncp[1] = 0; }
}

// ---------------------------------------------------------------------------
// Persistent recurrent kernel: 64 WGs x 128 threads = 256 wave32 tasks.
// All 4 waves of a WG share (dir, tj) and differ in batch tile ti, so the
// whole per-WG weight working set (the tj column slices, 96 KB f16) is staged
// once into LDS; the 2048-step loop reads B-fragments exclusively via DS.
// One grid sync per step (between LSTM h1-write and GRU h1-read) suffices.
// ---------------------------------------------------------------------------
__global__ __launch_bounds__(128) void brnn_persistent(
    const float* __restrict__ x,
    const float* __restrict__ b1x, const float* __restrict__ b1h,
    const float* __restrict__ b2x, const float* __restrict__ b2h,
    const _Float16* __restrict__ W1X, const _Float16* __restrict__ W1H,
    const _Float16* __restrict__ W2X, const _Float16* __restrict__ W2H,
    _Float16* __restrict__ H1H, float* __restrict__ C1,
    float* __restrict__ H2, _Float16* __restrict__ H2H,
    int* __restrict__ syncp) {
  const int lane = threadIdx.x & 31;
  const int wave = threadIdx.x >> 5;
  const int dir  = blockIdx.x >> 5;               // 32 WGs per direction
  const int wg   = blockIdx.x & 31;
  const int tj   = wg & 15;                       // hidden tile (WG-uniform)
  const int ti   = ((wg >> 4) << 2) + wave;       // batch tile 0..7
  const int ncol = tj * 16 + (lane & 15);         // hidden column this lane owns
  const int mrow = ti * 16 + ((lane >> 4) << 3);  // first of 8 rows this lane owns

  // ---- Stage this WG's weight column slices into LDS (96 KB) ----
  __shared__ _Float16 sW1X[4 * 16 * I_DIM];   // 16 KB
  __shared__ _Float16 sW1H[4 * 16 * H_DIM];   // 32 KB
  __shared__ _Float16 sW2X[3 * 16 * H_DIM];   // 24 KB
  __shared__ _Float16 sW2H[3 * 16 * H_DIM];   // 24 KB
  {
    const int tid = threadIdx.x;
    for (int i = tid; i < 4 * 16 * (I_DIM / 8); i += 128) {
      int row = i / (I_DIM / 8), kc8 = i % (I_DIM / 8);
      int g = row >> 4, r = row & 15;
      ((v8h*)sW1X)[i] =
          ((const v8h*)W1X)[((g * H_DIM + tj * 16 + r) * I_DIM) / 8 + kc8];
    }
    for (int i = tid; i < 4 * 16 * (H_DIM / 8); i += 128) {
      int row = i / (H_DIM / 8), kc8 = i % (H_DIM / 8);
      int g = row >> 4, r = row & 15;
      ((v8h*)sW1H)[i] =
          ((const v8h*)W1H)[((g * H_DIM + tj * 16 + r) * H_DIM) / 8 + kc8];
    }
    for (int i = tid; i < 3 * 16 * (H_DIM / 8); i += 128) {
      int row = i / (H_DIM / 8), kc8 = i % (H_DIM / 8);
      int g = row / 16, r = row % 16;
      ((v8h*)sW2X)[i] =
          ((const v8h*)W2X)[((g * H_DIM + tj * 16 + r) * H_DIM) / 8 + kc8];
      ((v8h*)sW2H)[i] =
          ((const v8h*)W2H)[((g * H_DIM + tj * 16 + r) * H_DIM) / 8 + kc8];
    }
    __syncthreads();
  }

  float* c1 = C1 + dir * BH;
  float* h2 = H2 + dir * BH;
  int* cnt = syncp;
  volatile int* gen = (volatile int*)(syncp + 1);

  // Hoist biases (column-only dependence)
  float bg[4];
  #pragma unroll
  for (int g = 0; g < 4; ++g) bg[g] = b1x[g * H_DIM + ncol] + b1h[g * H_DIM + ncol];
  float b2xv[3], b2hv[3];
  #pragma unroll
  for (int g = 0; g < 3; ++g) {
    b2xv[g] = b2x[g * H_DIM + ncol];
    b2hv[g] = b2h[g * H_DIM + ncol];
  }

  #pragma unroll 1
  for (int t = 0; t < T_DIM; ++t) {
    const int ts = dir ? (T_DIM - 1 - t) : t;
    const _Float16* h1_rd = H1H + ((t & 1) * 2 + dir) * BH;
    _Float16*       h1_wr = H1H + (((t + 1) & 1) * 2 + dir) * BH;
    const _Float16* h2_rd = H2H + ((t & 1) * 2 + dir) * BH;
    _Float16*       h2_wr = H2H + (((t + 1) & 1) * 2 + dir) * BH;

    // ================= Phase A: LSTM cell =================
    v8f acc0, acc1, acc2, acc3;
    #pragma unroll
    for (int r = 0; r < 8; ++r) {
      acc0[r] = bg[0]; acc1[r] = bg[1]; acc2[r] = bg[2]; acc3[r] = bg[3];
    }
    #pragma unroll
    for (int kc = 0; kc < I_DIM; kc += 32) {         // x_t @ W1x^T
      v16h a = frag_x(x, ts, ti * 16, kc, lane);
      if (t + 1 < T_DIM) prefetch_x(x, dir ? ts - 1 : ts + 1, ti * 16, kc, lane);
      acc0 = wmma32(a, frag_rm(sW1X, I_DIM, 0 * 16, kc, lane), acc0);
      acc1 = wmma32(a, frag_rm(sW1X, I_DIM, 1 * 16, kc, lane), acc1);
      acc2 = wmma32(a, frag_rm(sW1X, I_DIM, 2 * 16, kc, lane), acc2);
      acc3 = wmma32(a, frag_rm(sW1X, I_DIM, 3 * 16, kc, lane), acc3);
    }
    #pragma unroll
    for (int kc = 0; kc < H_DIM; kc += 32) {         // h1 @ W1h^T
      v16h a = frag_rm(h1_rd, H_DIM, ti * 16, kc, lane);
      acc0 = wmma32(a, frag_rm(sW1H, H_DIM, 0 * 16, kc, lane), acc0);
      acc1 = wmma32(a, frag_rm(sW1H, H_DIM, 1 * 16, kc, lane), acc1);
      acc2 = wmma32(a, frag_rm(sW1H, H_DIM, 2 * 16, kc, lane), acc2);
      acc3 = wmma32(a, frag_rm(sW1H, H_DIM, 3 * 16, kc, lane), acc3);
    }
    #pragma unroll
    for (int r = 0; r < 8; ++r) {                    // pointwise cell update
      int idx = (mrow + r) * H_DIM + ncol;
      float ig = sigm(acc0[r]);
      float fg = sigm(acc1[r]);
      float gg = tanhx(acc2[r]);
      float og = sigm(acc3[r]);
      float cv = c1[idx] * fg + ig * gg;
      c1[idx] = cv;
      h1_wr[idx] = (_Float16)(og * tanhx(cv));
    }

    grid_sync(cnt, gen);   // single device-wide sync per timestep

    // ================= Phase B: GRU cell =================
    v8f ax0, ax1, ax2, ah0, ah1, ah2;
    #pragma unroll
    for (int r = 0; r < 8; ++r) {
      ax0[r] = b2xv[0]; ax1[r] = b2xv[1]; ax2[r] = b2xv[2];
      ah0[r] = b2hv[0]; ah1[r] = b2hv[1]; ah2[r] = b2hv[2];
    }
    #pragma unroll
    for (int kc = 0; kc < H_DIM; kc += 32) {
      v16h a1 = frag_rm(h1_wr, H_DIM, ti * 16, kc, lane);  // new h1
      v16h a2 = frag_rm(h2_rd, H_DIM, ti * 16, kc, lane);  // old h2
      ax0 = wmma32(a1, frag_rm(sW2X, H_DIM, 0 * 16, kc, lane), ax0);
      ax1 = wmma32(a1, frag_rm(sW2X, H_DIM, 1 * 16, kc, lane), ax1);
      ax2 = wmma32(a1, frag_rm(sW2X, H_DIM, 2 * 16, kc, lane), ax2);
      ah0 = wmma32(a2, frag_rm(sW2H, H_DIM, 0 * 16, kc, lane), ah0);
      ah1 = wmma32(a2, frag_rm(sW2H, H_DIM, 1 * 16, kc, lane), ah1);
      ah2 = wmma32(a2, frag_rm(sW2H, H_DIM, 2 * 16, kc, lane), ah2);
    }
    #pragma unroll
    for (int r = 0; r < 8; ++r) {
      int idx = (mrow + r) * H_DIM + ncol;
      float rr = sigm(ax0[r] + ah0[r]);
      float zz = sigm(ax1[r] + ah1[r]);
      float nn = tanhx(ax2[r] + rr * ah2[r]);
      float hv = zz * h2[idx] + (1.0f - zz) * nn;
      h2[idx] = hv;
      h2_wr[idx] = (_Float16)hv;
    }
    // no second sync: next-step LSTM only touches h1 buffers already protected
    // by the next iteration's grid_sync (see dependence analysis).
  }
}

// ---------------------------------------------------------------------------
// Final FC: out[B,O] = [h_fwd | h_bwd] @ Wo^T + bo, WMMA, K = 512.
// 64 wave-tasks (8 batch tiles x 8 output tiles) -> 16 WGs x 128 threads.
// ---------------------------------------------------------------------------
__global__ __launch_bounds__(128) void brnn_fc(
    const _Float16* __restrict__ HF, const _Float16* __restrict__ HB,
    const _Float16* __restrict__ WO, const float* __restrict__ bo,
    float* __restrict__ out) {
  const int lane = threadIdx.x & 31;
  const int wave = threadIdx.x >> 5;
  const int task = blockIdx.x * 4 + wave;  // 0..63
  const int ti = task >> 3, tj = task & 7;
  const int ncol = tj * 16 + (lane & 15);
  const int mrow = ti * 16 + ((lane >> 4) << 3);

  v8f acc;
  float bv = bo[ncol];
  #pragma unroll
  for (int r = 0; r < 8; ++r) acc[r] = bv;

  #pragma unroll
  for (int kc = 0; kc < 2 * H_DIM; kc += 32) {
    const _Float16* src = (kc < H_DIM) ? HF : HB;
    v16h a = frag_rm(src, H_DIM, ti * 16, kc & (H_DIM - 1), lane);
    v16h b = frag_rm(WO, 2 * H_DIM, tj * 16, kc, lane);
    acc = wmma32(a, b, acc);
  }
  #pragma unroll
  for (int r = 0; r < 8; ++r) out[(mrow + r) * O_DIM + ncol] = acc[r];
}

// ---------------------------------------------------------------------------
extern "C" void kernel_launch(void* const* d_in, const int* in_sizes, int n_in,
                              void* d_out, int out_size, void* d_ws, size_t ws_size,
                              hipStream_t stream) {
  const float* x   = (const float*)d_in[0];
  const float* w1x = (const float*)d_in[1];
  const float* b1x = (const float*)d_in[2];
  const float* w1h = (const float*)d_in[3];
  const float* b1h = (const float*)d_in[4];
  const float* w2x = (const float*)d_in[5];
  const float* b2x = (const float*)d_in[6];
  const float* w2h = (const float*)d_in[7];
  const float* b2h = (const float*)d_in[8];
  const float* wo  = (const float*)d_in[9];
  const float* bo  = (const float*)d_in[10];
  float* out = (float*)d_out;

  // Workspace carve-out (256B-aligned slices), ~2.5 MB total.
  char* ws = (char*)d_ws;
  size_t off = 0;
  auto take = [&](size_t bytes) -> void* {
    void* p = ws + off;
    off = (off + bytes + 255) & ~(size_t)255;
    return p;
  };
  _Float16* W1X = (_Float16*)take((size_t)4 * H_DIM * I_DIM * 2);
  _Float16* W1H = (_Float16*)take((size_t)4 * H_DIM * H_DIM * 2);
  _Float16* W2X = (_Float16*)take((size_t)3 * H_DIM * H_DIM * 2);
  _Float16* W2H = (_Float16*)take((size_t)3 * H_DIM * H_DIM * 2);
  _Float16* WO  = (_Float16*)take((size_t)O_DIM * 2 * H_DIM * 2);
  _Float16* H1H = (_Float16*)take((size_t)4 * BH * 2);  // [parity][dir][B*H]
  _Float16* H2H = (_Float16*)take((size_t)4 * BH * 2);  // [parity][dir][B*H]
  float*    C1  = (float*)take((size_t)2 * BH * 4);     // [dir][B*H]
  float*    H2  = (float*)take((size_t)2 * BH * 4);     // [dir][B*H]
  int*      SYN = (int*)take(256);

  brnn_init<<<256, 256, 0, stream>>>(w1x, w1h, w2x, w2h, wo,
                                     W1X, W1H, W2X, W2H, WO,
                                     H1H, H2H, C1, H2, SYN);

  brnn_persistent<<<NWG, 128, 0, stream>>>(x, b1x, b1h, b2x, b2h,
                                           W1X, W1H, W2X, W2H,
                                           H1H, C1, H2, H2H, SYN);

  // Final state lands in parity-0 buffers (T even): [dir0 | dir1]
  brnn_fc<<<16, 128, 0, stream>>>(H2H, H2H + BH, WO, bo, out);
}
